// FINN_Burger_20864951124287
// MI455X (gfx1250) — matrix operand order repaired and checked by reading, not compiled, not run on hardware
//
#include <hip/hip_runtime.h>

typedef __attribute__((ext_vector_type(16))) _Float16 v16h;
typedef __attribute__((ext_vector_type(8)))  float    v8f;

#define H           512
#define KCHUNKS     16          // H / 32 (WMMA K depth)
#define NTILES      32          // H / 16 (WMMA N width)
#define ROWS_PER_WG 128
#define WAVES       8
#define THREADS     256

__device__ __forceinline__ float fast_tanh(float x) {
#if __has_builtin(__builtin_amdgcn_tanhf)
  return __builtin_amdgcn_tanhf(x);          // gfx1250 v_tanh_f32
#elif __has_builtin(__builtin_amdgcn_tanh_f32)
  return __builtin_amdgcn_tanh_f32(x);
#else
  return tanhf(x);
#endif
}

// Repack W2 (f32 [K=512][N=512] row-major) into f16 WMMA-B fragment-major layout.
// B 32x16 f16 k order used here (contiguous, same as A): lane = (k%32)/16*16 + n%16 ; half j = k%16.
// Fragment (kc,nt) is 32 lanes x 16 halves contiguous -> one 32B v16h load per lane.
__global__ void pack_w2_kernel(const float* __restrict__ w2,
                               _Float16* __restrict__ w2h) {
  int idx = blockIdx.x * blockDim.x + threadIdx.x;   // 0 .. H*H-1
  int k = idx / H, n = idx % H;
  int kc = k >> 5, kl = k & 31;
  int nt = n >> 4, nl = n & 15;
  int g  = kl >> 4, j = kl & 15;
  int lane = g * 16 + nl;
  size_t dst = (((size_t)nt * KCHUNKS + kc) * 32 + lane) * 16 + j;
  w2h[dst] = (_Float16)w2[idx];
}

// One Euler step: u_out = u_in + dt * flux(u_in), with a = MLP(u_in) via WMMA.
// A fragments (h1 = tanh(u*W1)) are held entirely in VGPRs (128 regs/lane);
// the GEMM inner loop issues only global B loads + v_wmma.
__global__ __launch_bounds__(THREADS) void finn_step_kernel(
    const float* __restrict__ u_in, float* __restrict__ u_out,
    const float* __restrict__ tarr, int step,
    const _Float16* __restrict__ w2h,
    const float* __restrict__ w1, const float* __restrict__ w3, int nx)
{
  __shared__ float w1s[H];
  __shared__ float w3s[H];
  __shared__ float aS[ROWS_PER_WG];

  const int tid  = threadIdx.x;
  const int wave = tid >> 5;
  const int lane = tid & 31;
  const int g    = lane >> 4;   // half-wave group
  const int nl   = lane & 15;

  for (int i = tid; i < H; i += THREADS) { w1s[i] = w1[i]; w3s[i] = w3[i]; }
  __syncthreads();

  const int rowBase = blockIdx.x * ROWS_PER_WG + wave * 16;
  const float u_r = u_in[rowBase + nl];   // this lane's row value (A row = lane%16)

  // ---- Layer 1: h1 = tanh(u * W1) built directly as register-resident
  //      WMMA A fragments.  16-bit A 16x32 k order used here (contiguous, matches B):
  //      lanes 0-15 -> K 0..15 ; lanes 16-31 -> K 16..31
  v16h afrag[KCHUNKS];
#pragma unroll
  for (int kc = 0; kc < KCHUNKS; ++kc) {
#pragma unroll
    for (int j = 0; j < 16; ++j) {
      const int kl = g * 16 + j;
      afrag[kc][j] = (_Float16)fast_tanh(u_r * w1s[kc * 32 + kl]);
    }
  }

  // ---- Layer 2 (WMMA GEMM over K=512) fused with layer-3 dot product ----
  float acc[8];
#pragma unroll
  for (int v = 0; v < 8; ++v) acc[v] = 0.f;

  for (int nt = 0; nt < NTILES; ++nt) {
    const float w3v = w3s[nt * 16 + nl];   // C col N = lane%16
    const _Float16* bbase =
        w2h + (((size_t)nt * KCHUNKS) * 32 + lane) * 16;
    v8f c = {};
#pragma unroll
    for (int kc = 0; kc < KCHUNKS; ++kc) {
      v16h b = *(const v16h*)(bbase + (size_t)kc * 32 * 16);
      c = __builtin_amdgcn_wmma_f32_16x16x32_f16(
              /*neg_a=*/false, afrag[kc], /*neg_b=*/false, b,
              /*c_mod=*/(short)0, c, /*reuse_a=*/false, /*reuse_b=*/false);
    }
#pragma unroll
    for (int v = 0; v < 8; ++v) acc[v] += fast_tanh(c[v]) * w3v;
  }

  // Reduce the 16 column-lanes of each half-wave; C row M = v + 8*(lane/16).
#pragma unroll
  for (int v = 0; v < 8; ++v) {
    float s = acc[v];
    s += __shfl_xor(s, 1, 16);
    s += __shfl_xor(s, 2, 16);
    s += __shfl_xor(s, 4, 16);
    s += __shfl_xor(s, 8, 16);
    acc[v] = s;
  }
  if (nl == 0) {
#pragma unroll
    for (int v = 0; v < 8; ++v)
      aS[wave * 16 + g * 8 + v] = fast_tanh(acc[v]);
  }
  __syncthreads();

  // ---- Flux + Euler update (S0=-1, S1=1, BC=0) ----
  if (tid < ROWS_PER_WG) {
    const int row = blockIdx.x * ROWS_PER_WG + tid;
    const float u  = u_in[row];
    const float ul = (row > 0)      ? u_in[row - 1] : 0.0f;   // BC0
    const float ur = (row < nx - 1) ? u_in[row + 1] : 0.0f;   // BC1
    const float a  = aS[tid];
    const float ap = fmaxf(a, 0.0f);
    const float am = fminf(a, 0.0f);
    const float D = 0.01f, invDX = 100.0f;
    const float flux = (ul - u) * (D + ap * invDX) + (ur - u) * (D - am * invDX);
    const float dt = tarr[step] - tarr[step - 1];
    u_out[row] = u + dt * flux;
  }
}

extern "C" void kernel_launch(void* const* d_in, const int* in_sizes, int n_in,
                              void* d_out, int out_size, void* d_ws, size_t ws_size,
                              hipStream_t stream) {
  const float* t  = (const float*)d_in[0];
  const float* u0 = (const float*)d_in[1];
  const float* w1 = (const float*)d_in[2];
  const float* w2 = (const float*)d_in[3];
  const float* w3 = (const float*)d_in[4];
  float* out = (float*)d_out;
  const int NT = in_sizes[0];
  const int nx = in_sizes[1];

  _Float16* w2h = (_Float16*)d_ws;   // 512 KB packed W2

  pack_w2_kernel<<<(H * H) / 256, 256, 0, stream>>>(w2, w2h);

  // pred[0] = u0
  hipMemcpyAsync(out, u0, (size_t)nx * sizeof(float),
                 hipMemcpyDeviceToDevice, stream);

  for (int s = 1; s < NT; ++s) {
    finn_step_kernel<<<nx / ROWS_PER_WG, THREADS, 0, stream>>>(
        out + (size_t)(s - 1) * nx,   // u_{s-1}
        out + (size_t)s * nx,         // u_s
        t, s, w2h, w1, w3, nx);
  }
}
